// Repr2Classifier_2877628088445
// MI455X (gfx1250) — compile-verified, matrix operand order checked by reading, not run on hardware
//
#include <hip/hip_runtime.h>

// ---------------- problem constants (match reference) ----------------
#define N_FLOW 100000
#define N_HOST 10000
#define E_HF   200000
#define E_FH   200000
#define E_FF   800000
#define FLOW_DIM 32
#define FD 35
#define HID 128
#define NCLS 10
#define NGRAPH 64
#define LAYERS 3

typedef __bf16 bf16_t;
typedef __attribute__((ext_vector_type(16))) __bf16 v16bf;
typedef __attribute__((ext_vector_type(8)))  __bf16 v8bf;
typedef __attribute__((ext_vector_type(8)))  float  v8f;

// round-to-nearest-even f32 -> bf16 bits (pure bit math, no libdevice)
__device__ inline unsigned short f2bfu(float f) {
    unsigned u = __float_as_uint(f);
    unsigned r = u + 0x7FFFu + ((u >> 16) & 1u);
    return (unsigned short)(r >> 16);
}
__device__ inline bf16_t f2bf(float f) {
    unsigned short h = f2bfu(f);
    return __builtin_bit_cast(bf16_t, h);
}

// ---------------- build padded flow features (f32, stride 128) ----------------
__global__ __launch_bounds__(128)
void build_xf_kernel(const float* __restrict__ x_flow,
                     const int* __restrict__ ports,
                     const int* __restrict__ flags,
                     const float* __restrict__ emb_port,
                     const float* __restrict__ emb_tcp,
                     float* __restrict__ xf) {
    int r = blockIdx.x;
    int c = threadIdx.x;
    float v = 0.0f;
    if (c < FLOW_DIM)      v = x_flow[(size_t)r * FLOW_DIM + c];
    else if (c == 32)      v = emb_port[ports[r]];
    else if (c == 33)      v = emb_tcp[flags[r] * 2 + 0];
    else if (c == 34)      v = emb_tcp[flags[r] * 2 + 1];
    xf[(size_t)r * HID + c] = v;
}

// ---------------- bulk f32 -> packed bf16 (4 elems / thread) -------------------
__global__ __launch_bounds__(256)
void cvt_f32_bf16_kernel(const float* __restrict__ src,
                         unsigned* __restrict__ dst, long long n4) {
    long long i = (long long)blockIdx.x * blockDim.x + threadIdx.x;
    if (i >= n4) return;
    float4 v = ((const float4*)src)[i];
    unsigned lo = (unsigned)f2bfu(v.x) | ((unsigned)f2bfu(v.y) << 16);
    unsigned hi = (unsigned)f2bfu(v.z) | ((unsigned)f2bfu(v.w) << 16);
    ((uint2*)dst)[i] = make_uint2(lo, hi);
}

// ---------------- weight staging: f32 [128 x K] -> bf16 [128 x 128] padded ----
__global__ __launch_bounds__(128)
void cvt_w_kernel(const float* __restrict__ src, bf16_t* __restrict__ dst, int K) {
    int r = blockIdx.x, c = threadIdx.x;
    float v = (c < K) ? src[(size_t)r * K + c] : 0.0f;
    dst[(size_t)r * HID + c] = f2bf(v);
}
// combined root weight: dst = src0 + src1 (same K), padded
__global__ __launch_bounds__(128)
void cvt_wsum_kernel(const float* __restrict__ s0, const float* __restrict__ s1,
                     bf16_t* __restrict__ dst, int K) {
    int r = blockIdx.x, c = threadIdx.x;
    float v = (c < K) ? (s0[(size_t)r * K + c] + s1[(size_t)r * K + c]) : 0.0f;
    dst[(size_t)r * HID + c] = f2bf(v);
}
// combined biases
__global__ __launch_bounds__(128)
void bias_stage_kernel(const float* __restrict__ b_hf, const float* __restrict__ b_ff,
                       const float* __restrict__ b_fh,
                       float* __restrict__ bias_nf, float* __restrict__ bias_hh) {
    int c = threadIdx.x;
    bias_nf[c] = b_hf[c] + b_ff[c];
    bias_hh[c] = b_fh[c];
}

// ---------------- edge scatter-add: agg[d] += src[s] (f32, bandwidth bound) ----
__global__ __launch_bounds__(256)
void scatter_add_kernel(const float* __restrict__ src,
                        const int* __restrict__ sidx,
                        const int* __restrict__ didx,
                        float* __restrict__ dst, int nedges) {
    long long idx = (long long)blockIdx.x * blockDim.x + threadIdx.x;
    int e = (int)(idx >> 5);
    int c = (int)(idx & 31) * 4;
    if (e >= nedges) return;
    const float4 v = *(const float4*)(src + (size_t)sidx[e] * HID + c);
    float* p = dst + (size_t)didx[e] * HID + c;
    atomicAdd(p + 0, v.x);
    atomicAdd(p + 1, v.y);
    atomicAdd(p + 2, v.z);
    atomicAdd(p + 3, v.w);
}

// ---------------- fused GraphConv GEMM ----------------------------------------
// C[Mx128] = sum_s A_s[Mx128pad]_bf16 @ W_s[128x128]_bf16^T + bias  (opt ReLU)
// block = 128 threads (4 waves); tile = 16 rows x 128 cols; each wave owns two
// 16x16 n-tiles. Weights for each source are staged into a 32KB LDS buffer via
// the CDNA5 async-to-LDS engine (global_load_async_to_lds_b128 + ASYNCcnt).
template <int NSRC, int RELU>
__global__ __launch_bounds__(128)
void fused_gconv_kernel(const bf16_t* __restrict__ A0,
                        const bf16_t* __restrict__ A1,
                        const bf16_t* __restrict__ A2,
                        const bf16_t* __restrict__ Wall,  // NSRC x [128x128] bf16
                        const float* __restrict__ bias,
                        float* __restrict__ C,
                        int Kp0, int Kp1, int Kp2) {
    __shared__ __attribute__((aligned(16))) bf16_t wlds[HID * HID];  // 32 KB

    const int lane    = threadIdx.x & 31;
    const int wave    = threadIdx.x >> 5;
    const int rowbase = blockIdx.x * 16;
    const int nlocal  = lane & 15;
    const int khalf   = lane >> 4;
    const int nb0 = wave * 32;
    const int nb1 = nb0 + 16;

    v8f acc0 = {};
    v8f acc1 = {};

    const unsigned lbase = (unsigned)(size_t)&wlds[0];

#pragma unroll
    for (int s = 0; s < NSRC; ++s) {
        const bf16_t* A  = (s == 0) ? A0 : ((s == 1) ? A1 : A2);
        const int     Kp = (s == 0) ? Kp0 : ((s == 1) ? Kp1 : Kp2);

        // ensure previous round's LDS consumers are done before overwrite
        __syncthreads();
        // cooperative async copy: 128 threads x 16 chunks x 16B = 32KB
        {
            const char* g = (const char*)(Wall + (size_t)s * HID * HID);
#pragma unroll
            for (int i = 0; i < (HID * HID * 2) / (128 * 16); ++i) {
                int chunk = i * 128 + threadIdx.x;
                unsigned loff = lbase + (unsigned)chunk * 16u;
                const char* gp = g + (size_t)chunk * 16u;
                asm volatile("global_load_async_to_lds_b128 %0, %1, off"
                             :: "v"(loff), "v"(gp) : "memory");
            }
            asm volatile("s_wait_asynccnt 0" ::: "memory");
            __syncthreads();
        }

        const bf16_t* arow = A + (size_t)(rowbase + nlocal) * HID;
        const bf16_t* w0row = &wlds[(size_t)(nb0 + nlocal) * HID];
        const bf16_t* w1row = &wlds[(size_t)(nb1 + nlocal) * HID];

        for (int k = 0; k < Kp; k += 32) {
            const int ka = k + khalf * 8;
            // A fragment: row = rowbase+nlocal, K = ka..ka+7 and ka+16..ka+23
            v8bf alo = *(const v8bf*)(arow + ka);
            v8bf ahi = *(const v8bf*)(arow + ka + 16);
            // B fragments from LDS: B[k][n] = W[n][k], contiguous along k
            v8bf b0l = *(const v8bf*)(w0row + ka);
            v8bf b0h = *(const v8bf*)(w0row + ka + 16);
            v8bf b1l = *(const v8bf*)(w1row + ka);
            v8bf b1h = *(const v8bf*)(w1row + ka + 16);
            v16bf af, bf0, bf1;
#pragma unroll
            for (int i = 0; i < 8; ++i) {
                af[i]  = alo[i]; af[i + 8]  = ahi[i];
                bf0[i] = b0l[i]; bf0[i + 8] = b0h[i];
                bf1[i] = b1l[i]; bf1[i + 8] = b1h[i];
            }
            acc0 = __builtin_amdgcn_wmma_f32_16x16x32_bf16(false, af, false, bf0,
                                                           (short)0, acc0, false, false);
            acc1 = __builtin_amdgcn_wmma_f32_16x16x32_bf16(false, af, false, bf1,
                                                           (short)0, acc1, false, false);
        }
    }

    // epilogue: C/D layout -> row = rowbase + g + 8*khalf, col = nbase + nlocal
#pragma unroll
    for (int t = 0; t < 2; ++t) {
        v8f acc = t ? acc1 : acc0;
        int col = (t ? nb1 : nb0) + nlocal;
        float bv = bias[col];
#pragma unroll
        for (int g = 0; g < 8; ++g) {
            int row = rowbase + g + khalf * 8;
            float v = acc[g] + bv;
            if (RELU) v = fmaxf(v, 0.0f);
            C[(size_t)row * HID + col] = v;
        }
    }
}

// ---------------- segment max pool via order-preserving uint atomicMax --------
__global__ __launch_bounds__(256)
void pool_max_kernel(const float* __restrict__ xf,
                     const int* __restrict__ batch,
                     unsigned* __restrict__ pooled, int n) {
    long long idx = (long long)blockIdx.x * blockDim.x + threadIdx.x;
    int r = (int)(idx >> 7);
    int c = (int)(idx & 127);
    if (r >= n) return;
    unsigned b = __float_as_uint(xf[(size_t)r * HID + c]);
    unsigned enc = b ^ (unsigned)(((int)b >> 31) | 0x80000000);
    atomicMax(&pooled[(size_t)batch[r] * HID + c], enc);
}

// ---------------- tiny MLP head: 64 graphs, 128->128->128->10 ------------------
__global__ __launch_bounds__(128)
void mlp_kernel(const unsigned* __restrict__ pooled,
                const float* __restrict__ W1, const float* __restrict__ b1,
                const float* __restrict__ W2, const float* __restrict__ b2,
                const float* __restrict__ W3, const float* __restrict__ b3,
                float* __restrict__ out) {
    __shared__ float s0[HID];
    __shared__ float s1[HID];
    int g = blockIdx.x, t = threadIdx.x;

    unsigned enc = pooled[(size_t)g * HID + t];
    unsigned mask = (enc & 0x80000000u) ? 0x80000000u : 0xFFFFFFFFu;
    s0[t] = __uint_as_float(enc ^ mask);
    __syncthreads();

    float acc = b1[t];
    for (int k = 0; k < HID; ++k) acc += W1[(size_t)t * HID + k] * s0[k];
    s1[t] = fmaxf(acc, 0.0f);
    __syncthreads();

    acc = b2[t];
    for (int k = 0; k < HID; ++k) acc += W2[(size_t)t * HID + k] * s1[k];
    __syncthreads();
    s0[t] = fmaxf(acc, 0.0f);
    __syncthreads();

    if (t < NCLS) {
        acc = b3[t];
        for (int k = 0; k < HID; ++k) acc += W3[(size_t)t * HID + k] * s0[k];
        out[(size_t)g * NCLS + t] = acc;
    }
}

// ---------------------------------------------------------------------------
extern "C" void kernel_launch(void* const* d_in, const int* in_sizes, int n_in,
                              void* d_out, int out_size, void* d_ws, size_t ws_size,
                              hipStream_t stream) {
    (void)in_sizes; (void)n_in; (void)out_size; (void)ws_size;

    // ---- inputs (setup_inputs dict order) ----
    const float* x_flow    = (const float*)d_in[0];
    const int*   dst_ports = (const int*)d_in[1];
    const int*   tcp_flags = (const int*)d_in[2];
    const int*   e_hf_src  = (const int*)d_in[3];
    const int*   e_hf_dst  = (const int*)d_in[4];
    const int*   e_fh_src  = (const int*)d_in[5];
    const int*   e_fh_dst  = (const int*)d_in[6];
    const int*   e_ff_src  = (const int*)d_in[7];
    const int*   e_ff_dst  = (const int*)d_in[8];
    const int*   flow_batch= (const int*)d_in[9];
    const float* emb_port  = (const float*)d_in[10];
    const float* emb_tcp   = (const float*)d_in[11];
    const float* W1 = (const float*)d_in[39];
    const float* b1 = (const float*)d_in[40];
    const float* W2 = (const float*)d_in[41];
    const float* b2 = (const float*)d_in[42];
    const float* W3 = (const float*)d_in[43];
    const float* b3 = (const float*)d_in[44];

    // ---- workspace layout ----
    char* ws = (char*)d_ws;
    size_t off = 0;
    auto take = [&](size_t bytes) -> void* {
        void* p = ws + off;
        off = (off + bytes + 255) & ~(size_t)255;
        return p;
    };
    float* xf   = (float*)take((size_t)N_FLOW * HID * 4);
    float* xh   = (float*)take((size_t)N_HOST * HID * 4);
    float* nf   = (float*)take((size_t)N_FLOW * HID * 4);
    float* hh   = (float*)take((size_t)N_HOST * HID * 4);
    float* aggA = (float*)take((size_t)N_FLOW * HID * 4);  // host->flow
    float* aggB = (float*)take((size_t)N_FLOW * HID * 4);  // flow->flow
    float* aggC = (float*)take((size_t)N_HOST * HID * 4);  // flow->host
    bf16_t* xf_bf   = (bf16_t*)take((size_t)N_FLOW * HID * 2);
    bf16_t* xh_bf   = (bf16_t*)take((size_t)N_HOST * HID * 2);
    bf16_t* aggA_bf = (bf16_t*)take((size_t)N_FLOW * HID * 2);
    bf16_t* aggB_bf = (bf16_t*)take((size_t)N_FLOW * HID * 2);
    bf16_t* aggC_bf = (bf16_t*)take((size_t)N_HOST * HID * 2);
    bf16_t* wbuf = (bf16_t*)take((size_t)5 * HID * HID * 2); // 5 staged weights
    float* bias_nf = (float*)take((size_t)HID * 4);
    float* bias_hh = (float*)take((size_t)HID * 4);
    unsigned* pooled = (unsigned*)take((size_t)NGRAPH * HID * 4);

    bf16_t* w_nf = wbuf;                           // slots 0..2: rel_hf, rel_ff, rootC
    bf16_t* w_hh = wbuf + (size_t)3 * HID * HID;   // slots 3..4: rel_fh, root_fh

    // ---- layer-0 features ----
    build_xf_kernel<<<N_FLOW, 128, 0, stream>>>(x_flow, dst_ports, tcp_flags,
                                                emb_port, emb_tcp, xf);
    hipMemsetAsync(xh, 0, (size_t)N_HOST * HID * 4, stream);

    const int sa_block = 256;
    auto sa_grid = [](long long edges) {
        return (unsigned)((edges * 32 + sa_block - 1) / sa_block);
    };
    auto c4_grid = [](long long n) {  // n elements, 4/thread, 256 threads/block
        return (unsigned)((n / 4 + 255) / 256);
    };
    auto kpad = [](int K) { return (K + 31) & ~31; };
    const long long nF = (long long)N_FLOW * HID;
    const long long nH = (long long)N_HOST * HID;

    for (int l = 0; l < LAYERS; ++l) {
        const int base = 12 + l * 9;
        const float* W_rel_hf  = (const float*)d_in[base + 0];
        const float* b_rel_hf  = (const float*)d_in[base + 1];
        const float* W_root_hf = (const float*)d_in[base + 2];
        const float* W_rel_fh  = (const float*)d_in[base + 3];
        const float* b_rel_fh  = (const float*)d_in[base + 4];
        const float* W_root_fh = (const float*)d_in[base + 5];
        const float* W_rel_ff  = (const float*)d_in[base + 6];
        const float* b_rel_ff  = (const float*)d_in[base + 7];
        const float* W_root_ff = (const float*)d_in[base + 8];

        const int K_rel_hf  = l ? HID : 1;
        const int K_root    = l ? HID : FD;   // root_hf and root_ff share K
        const int K_rel_fh  = l ? HID : FD;
        const int K_root_fh = l ? HID : 1;
        const int K_rel_ff  = l ? HID : FD;

        // stage weights (bf16, zero-padded to 128x128) + combined bias
        cvt_w_kernel   <<<HID, HID, 0, stream>>>(W_rel_hf,  w_nf + 0 * HID * HID, K_rel_hf);
        cvt_w_kernel   <<<HID, HID, 0, stream>>>(W_rel_ff,  w_nf + 1 * HID * HID, K_rel_ff);
        cvt_wsum_kernel<<<HID, HID, 0, stream>>>(W_root_hf, W_root_ff,
                                                 w_nf + 2 * HID * HID, K_root);
        cvt_w_kernel   <<<HID, HID, 0, stream>>>(W_rel_fh,  w_hh + 0 * HID * HID, K_rel_fh);
        cvt_w_kernel   <<<HID, HID, 0, stream>>>(W_root_fh, w_hh + 1 * HID * HID, K_root_fh);
        bias_stage_kernel<<<1, HID, 0, stream>>>(b_rel_hf, b_rel_ff, b_rel_fh,
                                                 bias_nf, bias_hh);

        // bf16 activations for the GEMMs
        cvt_f32_bf16_kernel<<<c4_grid(nF), 256, 0, stream>>>(xf, (unsigned*)xf_bf, nF / 4);
        cvt_f32_bf16_kernel<<<c4_grid(nH), 256, 0, stream>>>(xh, (unsigned*)xh_bf, nH / 4);

        // edge aggregation (gather -> atomic scatter-add), then pack to bf16
        hipMemsetAsync(aggA, 0, (size_t)N_FLOW * HID * 4, stream);
        hipMemsetAsync(aggB, 0, (size_t)N_FLOW * HID * 4, stream);
        hipMemsetAsync(aggC, 0, (size_t)N_HOST * HID * 4, stream);
        scatter_add_kernel<<<sa_grid(E_HF), sa_block, 0, stream>>>(xh, e_hf_src, e_hf_dst, aggA, E_HF);
        scatter_add_kernel<<<sa_grid(E_FF), sa_block, 0, stream>>>(xf, e_ff_src, e_ff_dst, aggB, E_FF);
        scatter_add_kernel<<<sa_grid(E_FH), sa_block, 0, stream>>>(xf, e_fh_src, e_fh_dst, aggC, E_FH);
        cvt_f32_bf16_kernel<<<c4_grid(nF), 256, 0, stream>>>(aggA, (unsigned*)aggA_bf, nF / 4);
        cvt_f32_bf16_kernel<<<c4_grid(nF), 256, 0, stream>>>(aggB, (unsigned*)aggB_bf, nF / 4);
        cvt_f32_bf16_kernel<<<c4_grid(nH), 256, 0, stream>>>(aggC, (unsigned*)aggC_bf, nH / 4);

        const unsigned gF = N_FLOW / 16, gH = N_HOST / 16;
        // NF: aggA@rel_hf^T + aggB@rel_ff^T + xf@(root_hf+root_ff)^T + (b_hf+b_ff)
        if (l < LAYERS - 1) {
            fused_gconv_kernel<3, 1><<<gF, 128, 0, stream>>>(
                aggA_bf, aggB_bf, xf_bf, w_nf, bias_nf, nf,
                kpad(K_rel_hf), kpad(K_rel_ff), kpad(K_root));
            fused_gconv_kernel<2, 1><<<gH, 128, 0, stream>>>(
                aggC_bf, xh_bf, nullptr, w_hh, bias_hh, hh,
                kpad(K_rel_fh), kpad(K_root_fh), 0);
        } else {
            fused_gconv_kernel<3, 0><<<gF, 128, 0, stream>>>(
                aggA_bf, aggB_bf, xf_bf, w_nf, bias_nf, nf,
                kpad(K_rel_hf), kpad(K_rel_ff), kpad(K_root));
            fused_gconv_kernel<2, 0><<<gH, 128, 0, stream>>>(
                aggC_bf, xh_bf, nullptr, w_hh, bias_hh, hh,
                kpad(K_rel_fh), kpad(K_root_fh), 0);
        }

        // swap roles (deterministic pointer swap)
        float* t;
        t = xf; xf = nf; nf = t;
        t = xh; xh = hh; hh = t;
    }

    // ---- global max pool + MLP head ----
    hipMemsetAsync(pooled, 0, (size_t)NGRAPH * HID * 4, stream);
    {
        long long threads = (long long)N_FLOW * HID;
        unsigned grid = (unsigned)((threads + 255) / 256);
        pool_max_kernel<<<grid, 256, 0, stream>>>(xf, flow_batch, pooled, N_FLOW);
    }
    mlp_kernel<<<NGRAPH, 128, 0, stream>>>(pooled, W1, b1, W2, b2, W3, b3,
                                           (float*)d_out);
}
